// CrossAttentionFusion_61761629716762
// MI455X (gfx1250) — matrix operand order repaired and checked by reading, not compiled, not run on hardware
//
#include <hip/hip_runtime.h>

// ---------------------------------------------------------------------------
// CDNA5 / gfx1250 fused CrossAttentionFusion forward.
// Math simplifications (exact): softmax over 1 key == 1.0 -> Q/K dead;
// attn_x = LN(other) @ (out_w@Wv)^T + (out_w@bv + out_b).
// All GEMMs via v_wmma_f32_16x16x32_bf16, LN stats in f32.
// ---------------------------------------------------------------------------

typedef __attribute__((ext_vector_type(16))) __bf16         v16bf;
typedef __attribute__((ext_vector_type(8)))  float          v8f;
typedef __attribute__((ext_vector_type(8)))  unsigned short v8u;

// A fragment (16x32 bf16, MxK): lane l holds row m=l&15;
// half-wave 0: K = {kt..kt+7, kt+16..kt+23}; half-wave 1: +8.
static __device__ __forceinline__ v16bf load_a_frag(const __bf16* p) {
  v16bf r;
  ((v8u*)&r)[0] = *(const v8u*)(p);        // 8 bf16, 16B aligned
  ((v8u*)&r)[1] = *(const v8u*)(p + 16);   // next 8 bf16 at K+16
  return r;
}
// B fragment (32x16 bf16, KxN): lane l holds col n=l&15;
// half-wave 0: K = {kt..kt+7, kt+16..kt+23}; half-wave 1: +8.
static __device__ __forceinline__ v16bf load_b_frag(const __bf16* p) {
  v16bf r;
  ((v8u*)&r)[0] = *(const v8u*)(p);
  ((v8u*)&r)[1] = *(const v8u*)(p + 16);
  return r;
}

#define WMMA_BF16(a, b, c) \
  __builtin_amdgcn_wmma_f32_16x16x32_bf16(false, (a), false, (b), (short)0, (c), false, false)

// ---------------------------------------------------------------------------
// Prelaunch helpers
// ---------------------------------------------------------------------------
__global__ void cvt_bf16(const float* __restrict__ in, __bf16* __restrict__ out, int n) {
  int i = blockIdx.x * 256 + threadIdx.x;
  if (i < n) out[i] = (__bf16)in[i];
}

// Wc[i][j] = sum_k out_w[i][k] * Wv[k][j],  Wv = in_proj_w rows [512:768)
// bc[i]    = out_b[i] + sum_k out_w[i][k] * in_proj_b[512+k]
__global__ void build_wc(const float* __restrict__ out_w,
                         const float* __restrict__ in_proj_w,
                         const float* __restrict__ in_proj_b,
                         const float* __restrict__ out_b,
                         __bf16* __restrict__ Wc, float* __restrict__ bc) {
  const int i = blockIdx.x;    // output row (256)
  const int j = threadIdx.x;   // output col (256)
  float acc = 0.f;
  for (int k = 0; k < 256; ++k)
    acc += out_w[i * 256 + k] * in_proj_w[(512 + k) * 256 + j];
  Wc[i * 256 + j] = (__bf16)acc;
  if (j == 0) {
    float bb = out_b[i];
    for (int k = 0; k < 256; ++k) bb += out_w[i * 256 + k] * in_proj_b[512 + k];
    bc[i] = bb;
  }
}

// ---------------------------------------------------------------------------
// Block-wide LayerNorm over 16 rows x 256 cols held in sX (f32).
// 16 threads per row; optional bf16 and/or f32 output, optional ReLU.
// ---------------------------------------------------------------------------
static __device__ void block_ln(const float* sX, const float* __restrict__ g,
                                const float* __restrict__ bta,
                                __bf16* obf, float* of32, bool relu,
                                float* sRed, float* sStat) {
  const int t = threadIdx.x;
  const int r = t >> 4, j = t & 15;
  float s0 = 0.f, s1 = 0.f;
  for (int c = j; c < 256; c += 16) {
    float v = sX[r * 256 + c];
    s0 += v; s1 += v * v;
  }
  sRed[r * 16 + j]       = s0;
  sRed[256 + r * 16 + j] = s1;
  __syncthreads();
  if (t < 16) {
    float a0 = 0.f, a1 = 0.f;
    for (int q = 0; q < 16; ++q) { a0 += sRed[t * 16 + q]; a1 += sRed[256 + t * 16 + q]; }
    float mean = a0 * (1.f / 256.f);
    float var  = a1 * (1.f / 256.f) - mean * mean;
    sStat[t]      = mean;
    sStat[16 + t] = rsqrtf(var + 1e-5f);
  }
  __syncthreads();
  const float mean = sStat[r], rstd = sStat[16 + r];
  for (int c = j; c < 256; c += 16) {
    float y = (sX[r * 256 + c] - mean) * rstd * g[c] + bta[c];
    if (relu) y = fmaxf(y, 0.f);
    if (obf)  obf[r * 256 + c]  = (__bf16)y;
    if (of32) of32[r * 256 + c] = y;
  }
  __syncthreads();
}

// ---------------------------------------------------------------------------
// Fused forward: 16 rows per block, 8 waves split N into 2 tiles each.
// ---------------------------------------------------------------------------
__global__ __launch_bounds__(256)
void fused_forward(const float* __restrict__ bio,  const float* __restrict__ text,
                   const __bf16* __restrict__ bio_w,  const float* __restrict__ bio_b,
                   const __bf16* __restrict__ text_w, const float* __restrict__ text_b,
                   const float* __restrict__ ln_bio_g,  const float* __restrict__ ln_bio_b,
                   const float* __restrict__ ln_text_g, const float* __restrict__ ln_text_b,
                   const __bf16* __restrict__ Wc, const float* __restrict__ bc,
                   const __bf16* __restrict__ cls1_w, const float* __restrict__ cls1_b,
                   const float* __restrict__ cls_ln_g, const float* __restrict__ cls_ln_b,
                   const float* __restrict__ cls2_w, const float* __restrict__ cls2_b,
                   float* __restrict__ out, int Btot) {
  __shared__ __align__(16) __bf16 sAt[16 * 768];    // text tile (bf16)
  __shared__ __align__(16) __bf16 sAb[16 * 32];     // bio tile
  __shared__ __align__(16) float  sPre[16 * 256];   // pre-LN scratch (reused 3x)
  __shared__ __align__(16) __bf16 sXt[16 * 256];    // LN(text proj)
  __shared__ __align__(16) __bf16 sXb[16 * 256];    // LN(bio proj)
  __shared__ __align__(16) __bf16 sFused[16 * 512]; // [attn_bio | attn_text]
  __shared__ __align__(16) float  sH[16 * 256];     // relu(LN(cls1)) f32
  __shared__ float sRed[512];
  __shared__ float sStat[32];

  const int t    = threadIdx.x;
  const int lane = t & 31;
  const int wave = t >> 5;            // 0..7
  const int m    = lane & 15;         // A row / B col within tile
  const int hi   = lane >> 4;         // half-wave
  const int kA   = hi * 8;            // A-frag K sub-offset
  const int kB   = hi * 8;            // B-frag K sub-offset
  const int row0 = blockIdx.x * 16;

  // ---- stage inputs to LDS (coalesced f32 reads, bf16 stores) ----
  for (int i = t; i < 16 * 768; i += 256) {
    int r = i / 768, c = i - r * 768;
    int gr = row0 + r;
    float x = (gr < Btot) ? text[(size_t)gr * 768 + c] : 0.f;
    sAt[i] = (__bf16)x;
  }
  for (int i = t; i < 16 * 32; i += 256) {
    int gr = row0 + (i >> 5);
    float x = (gr < Btot) ? bio[(size_t)gr * 32 + (i & 31)] : 0.f;
    sAb[i] = (__bf16)x;
  }
  __syncthreads();

  // ---- phase 1a: text projection  [16x768] @ [768x256]^T ----
  for (int s = 0; s < 2; ++s) {
    const int ncol = (wave * 2 + s) * 16;
    v8f acc = {0.f, 0.f, 0.f, 0.f, 0.f, 0.f, 0.f, 0.f};
    for (int kt = 0; kt < 768; kt += 32) {
      v16bf a = load_a_frag(sAt + m * 768 + kt + kA);
      v16bf b = load_b_frag(text_w + (ncol + m) * 768 + kt + kB);
      acc = WMMA_BF16(a, b, acc);
    }
    const float bias = text_b[ncol + m];
#pragma unroll
    for (int r = 0; r < 8; ++r)
      sPre[(r + 8 * hi) * 256 + ncol + m] = acc[r] + bias;
  }
  __syncthreads();
  block_ln(sPre, ln_text_g, ln_text_b, sXt, nullptr, false, sRed, sStat);

  // ---- phase 1b: bio projection  [16x32] @ [32x256]^T ----
  for (int s = 0; s < 2; ++s) {
    const int ncol = (wave * 2 + s) * 16;
    v8f acc = {0.f, 0.f, 0.f, 0.f, 0.f, 0.f, 0.f, 0.f};
    v16bf a = load_a_frag(sAb + m * 32 + kA);
    v16bf b = load_b_frag(bio_w + (ncol + m) * 32 + kB);
    acc = WMMA_BF16(a, b, acc);
    const float bias = bio_b[ncol + m];
#pragma unroll
    for (int r = 0; r < 8; ++r)
      sPre[(r + 8 * hi) * 256 + ncol + m] = acc[r] + bias;
  }
  __syncthreads();
  block_ln(sPre, ln_bio_g, ln_bio_b, sXb, nullptr, false, sRed, sStat);

  // ---- phase 2: collapsed attention  attn = LN(other) @ Wc^T + bc ----
  for (int s = 0; s < 2; ++s) {
    const int ncol = (wave * 2 + s) * 16;
    v8f accB = {0.f, 0.f, 0.f, 0.f, 0.f, 0.f, 0.f, 0.f};  // attn_bio  (from text)
    v8f accT = {0.f, 0.f, 0.f, 0.f, 0.f, 0.f, 0.f, 0.f};  // attn_text (from bio)
    for (int kt = 0; kt < 256; kt += 32) {
      v16bf b  = load_b_frag(Wc + (ncol + m) * 256 + kt + kB);
      v16bf at = load_a_frag(sXt + m * 256 + kt + kA);
      v16bf ab = load_a_frag(sXb + m * 256 + kt + kA);
      accB = WMMA_BF16(at, b, accB);
      accT = WMMA_BF16(ab, b, accT);
    }
    const float bias = bc[ncol + m];
#pragma unroll
    for (int r = 0; r < 8; ++r) {
      const int rr = r + 8 * hi;
      sFused[rr * 512 + ncol + m]       = (__bf16)(accB[r] + bias);
      sFused[rr * 512 + 256 + ncol + m] = (__bf16)(accT[r] + bias);
    }
  }
  __syncthreads();

  // ---- phase 3: cls1  [16x512] @ [512x256]^T ----
  for (int s = 0; s < 2; ++s) {
    const int ncol = (wave * 2 + s) * 16;
    v8f acc = {0.f, 0.f, 0.f, 0.f, 0.f, 0.f, 0.f, 0.f};
    for (int kt = 0; kt < 512; kt += 32) {
      v16bf a = load_a_frag(sFused + m * 512 + kt + kA);
      v16bf b = load_b_frag(cls1_w + (ncol + m) * 512 + kt + kB);
      acc = WMMA_BF16(a, b, acc);
    }
    const float bias = cls1_b[ncol + m];
#pragma unroll
    for (int r = 0; r < 8; ++r)
      sPre[(r + 8 * hi) * 256 + ncol + m] = acc[r] + bias;
  }
  __syncthreads();
  block_ln(sPre, cls_ln_g, cls_ln_b, nullptr, sH, true, sRed, sStat);

  // ---- phase 4: cls2 (only 2 outputs/row) -> plain f32 dot ----
  if (t < 32) {
    const int r = t >> 1, c = t & 1;
    const int gr = row0 + r;
    if (gr < Btot) {
      float acc = cls2_b[c];
      for (int k = 0; k < 256; ++k) acc += sH[r * 256 + k] * cls2_w[c * 256 + k];
      out[(size_t)gr * 2 + c] = acc;
    }
  }
}

// ---------------------------------------------------------------------------
// Launch
// ---------------------------------------------------------------------------
extern "C" void kernel_launch(void* const* d_in, const int* in_sizes, int n_in,
                              void* d_out, int out_size, void* d_ws, size_t ws_size,
                              hipStream_t stream) {
  const float* bio       = (const float*)d_in[0];
  const float* text      = (const float*)d_in[1];
  const float* bio_w     = (const float*)d_in[2];
  const float* bio_b     = (const float*)d_in[3];
  const float* text_w    = (const float*)d_in[4];
  const float* text_b    = (const float*)d_in[5];
  const float* ln_bio_g  = (const float*)d_in[6];
  const float* ln_bio_b  = (const float*)d_in[7];
  const float* ln_text_g = (const float*)d_in[8];
  const float* ln_text_b = (const float*)d_in[9];
  const float* in_proj_w = (const float*)d_in[10];
  const float* in_proj_b = (const float*)d_in[11];
  const float* out_w     = (const float*)d_in[12];
  const float* out_b     = (const float*)d_in[13];
  const float* cls1_w    = (const float*)d_in[14];
  const float* cls1_b    = (const float*)d_in[15];
  const float* cls_ln_g  = (const float*)d_in[16];
  const float* cls_ln_b  = (const float*)d_in[17];
  const float* cls2_w    = (const float*)d_in[18];
  const float* cls2_b    = (const float*)d_in[19];

  char* ws = (char*)d_ws;
  __bf16* text_w_bf = (__bf16*)(ws + 0);        // 256*768*2 = 393216
  __bf16* cls1_w_bf = (__bf16*)(ws + 393216);   // 256*512*2 = 262144
  __bf16* bio_w_bf  = (__bf16*)(ws + 655360);   // 256*32*2  = 16384
  __bf16* Wc        = (__bf16*)(ws + 671744);   // 256*256*2 = 131072
  float*  bc        = (float*) (ws + 802816);   // 256*4

  cvt_bf16<<<768, 256, 0, stream>>>(text_w, text_w_bf, 256 * 768);
  cvt_bf16<<<512, 256, 0, stream>>>(cls1_w, cls1_w_bf, 256 * 512);
  cvt_bf16<<<32,  256, 0, stream>>>(bio_w,  bio_w_bf,  256 * 32);
  build_wc<<<256, 256, 0, stream>>>(out_w, in_proj_w, in_proj_b, out_b, Wc, bc);

  const int Btot = in_sizes[0] / 32;  // bio is [B,32]
  fused_forward<<<(Btot + 15) / 16, 256, 0, stream>>>(
      bio, text, bio_w_bf, bio_b, text_w_bf, text_b,
      ln_bio_g, ln_bio_b, ln_text_g, ln_text_b,
      Wc, bc, cls1_w_bf, cls1_b, cls_ln_g, cls_ln_b,
      cls2_w, cls2_b, (float*)d_out, Btot);
}
